// MHSEAtt_26886495273531
// MI455X (gfx1250) — compile-verified, matrix operand order checked by reading, not run on hardware
//
#include <hip/hip_runtime.h>

// ---------------------------------------------------------------------------
// MI455X (gfx1250) fused MHA: 3 proj GEMMs -> flash attention -> out GEMM
// Matrix math: v_wmma_f32_16x16x32_bf16. Data movement: TDM tensor_load_to_lds
// (guarded) -> async global->LDS (guarded) -> plain loads, in that preference.
// ---------------------------------------------------------------------------

typedef __attribute__((ext_vector_type(16))) __bf16 v16bf;
typedef __attribute__((ext_vector_type(8)))  __bf16 v8bf;
typedef __attribute__((ext_vector_type(4)))  __bf16 v4bf;
typedef __attribute__((ext_vector_type(8)))  float  v8f;
typedef unsigned int u32x4 __attribute__((ext_vector_type(4)));
typedef int          i32x8 __attribute__((ext_vector_type(8)));
typedef int          i32x4 __attribute__((ext_vector_type(4)));

#if __has_builtin(__builtin_amdgcn_tensor_load_to_lds)
#define HAVE_TDM 1
#else
#define HAVE_TDM 0
#endif
#if __has_builtin(__builtin_amdgcn_global_load_async_to_lds_b128)
#define HAVE_ASYNC 1
#else
#define HAVE_ASYNC 0
#endif

constexpr int Bc  = 4;
constexpr int Sc  = 1024;
constexpr int Dc  = 1024;
constexpr int Hc  = 16;
constexpr int DHc = 64;
constexpr int Mtot = Bc * Sc;   // 4096

__device__ inline v8f zero8() {
    v8f z;
#pragma unroll
    for (int i = 0; i < 8; ++i) z[i] = 0.0f;
    return z;
}

__device__ inline unsigned lds_offset(const void* p) {
    return (unsigned)(size_t)((__attribute__((address_space(3))) const void*)p);
}

#if HAVE_TDM
// Issue one 2-D bf16 TDM tile load: tile_w x tile_h elements, global row
// stride `gstride` (elements), LDS rows padded per pad_int/pad_amt codes.
// pad_interval code: 0=2,1=4,2=8,3=16,4=32.. DWORDs; pad_amount code: n -> n+1 DWORDs.
__device__ inline void tdm_load_2d_bf16(const void* gsrc, unsigned lds_off,
                                        unsigned tile_w, unsigned tile_h,
                                        unsigned gstride,
                                        unsigned tdim0, unsigned tdim1,
                                        int pad_int_code, int pad_amt_code) {
    unsigned long long ga = (unsigned long long)(size_t)gsrc;
    u32x4 g0;
    g0[0] = 1u;                                        // count=1 (valid), user mode
    g0[1] = lds_off;                                   // lds_addr
    g0[2] = (unsigned)(ga & 0xFFFFFFFFu);              // global_addr[31:0]
    g0[3] = (unsigned)((ga >> 32) & 0x01FFFFFFu)       // global_addr[56:32]
          | (2u << 30);                                // type = 2 ("image")
    i32x8 g1;
    g1[0] = (int)((1u << 16)                           // data_size = 2 bytes
          | (1u << 20)                                 // pad_enable
          | ((unsigned)pad_int_code << 22)
          | ((unsigned)pad_amt_code << 25));
    g1[1] = (int)((tdim0 & 0xFFFFu) << 16);            // tensor_dim0[15:0] @ bits 63:48
    g1[2] = (int)(((tdim0 >> 16) & 0xFFFFu)            // tensor_dim0[31:16]
          | ((tdim1 & 0xFFFFu) << 16));                // tensor_dim1[15:0]
    g1[3] = (int)(((tdim1 >> 16) & 0xFFFFu)
          | ((tile_w & 0xFFFFu) << 16));               // tile_dim0
    g1[4] = (int)(tile_h & 0xFFFFu);                   // tile_dim1 (tile_dim2 = 0)
    g1[5] = (int)gstride;                              // tensor_dim0_stride[31:0]
    g1[6] = 0;                                         // stride hi + dim1_stride lo
    g1[7] = 0;
    i32x4 z4 = {0, 0, 0, 0};
    i32x8 z8 = {0, 0, 0, 0, 0, 0, 0, 0};
    // 6-arg form (clang on this toolchain): g0, g1, g2, g3, extra, cpol
    __builtin_amdgcn_tensor_load_to_lds(g0, g1, z4, z4, z8, 0);
}
#endif

// 16x32 bf16 A/B fragment from an LDS row (ISA 7.12.2 layout):
// lane L holds row M=L%16; lanes<16 take K {0..7,16..23}, lanes>=16 {8..15,24..31}.
__device__ inline v16bf ld_frag(const __bf16* rowptr, int kb) {
    v8bf lo = *(const v8bf*)(rowptr + kb);
    v8bf hi = *(const v8bf*)(rowptr + kb + 16);
    v16bf f;
#pragma unroll
    for (int i = 0; i < 8; ++i) { f[i] = lo[i]; f[i + 8] = hi[i]; }
    return f;
}

// ---------------------------------------------------------------------------
// GEMM: Out[M,N] = A[M,K] * W[N,K]^T + bias  (M=4096, N=K=1024)
// 256 threads = 8 waves; block tile 128x128; wave tile 64x32 (8 WMMA / K-step).
// A_BF16=1: A is bf16 (ctx) -> staged via TDM when available.
// ---------------------------------------------------------------------------
template <int A_BF16>
__global__ __launch_bounds__(256)
void gemm128_wmma_bf16(const void* __restrict__ Ap,
                       const float* __restrict__ W,
                       const float* __restrict__ bias,
                       void* __restrict__ Outp, int split) {
    constexpr int K = Dc, N = Dc;
    constexpr bool a_tdm = (A_BF16 != 0) && (HAVE_TDM != 0);
    __shared__ __bf16 As[128][40];   // 32 cols + 8 pad -> 80B row stride (16B aligned)
    __shared__ __bf16 Bs[128][40];

    const int tid  = threadIdx.x;
    const int lane = tid & 31;
    const int wave = tid >> 5;
    const int wm   = wave & 1;
    const int wn   = wave >> 1;
    const int bm   = blockIdx.y * 128;
    const int bn   = blockIdx.x * 128;
    const int r15  = lane & 15;
    const int kb   = (lane < 16) ? 0 : 8;

    const float*  Af = (const float*)Ap;
    const __bf16* Ab = (const __bf16*)Ap;

    v8f acc[4][2];
#pragma unroll
    for (int mi = 0; mi < 4; ++mi)
#pragma unroll
        for (int ni = 0; ni < 2; ++ni) acc[mi][ni] = zero8();

    float4 wreg[4];
    float4 aregf[4];
    v4bf   aregb[4];

    auto load_stage = [&](int kk) {
#pragma unroll
        for (int it = 0; it < 4; ++it) {
            int idx = it * 256 + tid;       // 0..1023, 4 elements each
            int r   = idx >> 3;
            int c4  = idx & 7;
            int gc  = kk + c4 * 4;
            if constexpr (A_BF16) {
                if constexpr (!a_tdm)
                    aregb[it] = *(const v4bf*)(Ab + (size_t)(bm + r) * K + gc);
            } else {
                aregf[it] = *(const float4*)(Af + (size_t)(bm + r) * K + gc);
            }
            wreg[it] = *(const float4*)(W + (size_t)(bn + r) * K + gc);
        }
    };
    auto store_stage = [&]() {
#pragma unroll
        for (int it = 0; it < 4; ++it) {
            int idx = it * 256 + tid;
            int r   = idx >> 3;
            int c4  = idx & 7;
            if constexpr (A_BF16) {
                if constexpr (!a_tdm) *(v4bf*)&As[r][c4 * 4] = aregb[it];
            } else {
                v4bf t;
                t[0] = (__bf16)aregf[it].x; t[1] = (__bf16)aregf[it].y;
                t[2] = (__bf16)aregf[it].z; t[3] = (__bf16)aregf[it].w;
                *(v4bf*)&As[r][c4 * 4] = t;
            }
            v4bf w;
            w[0] = (__bf16)wreg[it].x; w[1] = (__bf16)wreg[it].y;
            w[2] = (__bf16)wreg[it].z; w[3] = (__bf16)wreg[it].w;
            *(v4bf*)&Bs[r][c4 * 4] = w;
        }
    };

    load_stage(0);
    for (int kk = 0; kk < K; kk += 32) {
        __syncthreads();                       // previous iter's LDS reads done
#if HAVE_TDM
        if (A_BF16 && tid == 0) {
            // 128x32 bf16 tile, global row stride 1024; LDS rows padded:
            // 32 elems = 16 DW -> pad_interval code 3; pad 8 elems = 4 DW -> code 3
            tdm_load_2d_bf16(Ab + (size_t)bm * K + kk, lds_offset(&As[0][0]),
                             32u, 128u, (unsigned)K, (unsigned)K, (unsigned)Mtot,
                             3, 3);
        }
#endif
        store_stage();
#if HAVE_TDM
        if (A_BF16 && tid == 0) __builtin_amdgcn_s_wait_tensorcnt(0);
#endif
        __syncthreads();
        if (kk + 32 < K) {
            load_stage(kk + 32);               // next slab in flight during WMMA
            __builtin_prefetch((const char*)(W + (size_t)(bn + (tid >> 3)) * K + kk + 64), 0, 1);
        }

        v16bf af[4], bf[2];
#pragma unroll
        for (int mi = 0; mi < 4; ++mi)
            af[mi] = ld_frag(&As[wm * 64 + mi * 16 + r15][0], kb);
#pragma unroll
        for (int ni = 0; ni < 2; ++ni)
            bf[ni] = ld_frag(&Bs[wn * 32 + ni * 16 + r15][0], kb);
#pragma unroll
        for (int mi = 0; mi < 4; ++mi)
#pragma unroll
            for (int ni = 0; ni < 2; ++ni)
                acc[mi][ni] = __builtin_amdgcn_wmma_f32_16x16x32_bf16(
                    false, af[mi], false, bf[ni], (short)0, acc[mi][ni], false, false);
    }

#pragma unroll
    for (int mi = 0; mi < 4; ++mi) {
#pragma unroll
        for (int ni = 0; ni < 2; ++ni) {
            int col = bn + wn * 32 + ni * 16 + r15;
            float bv = bias[col];
#pragma unroll
            for (int i = 0; i < 8; ++i) {
                int row = bm + wm * 64 + mi * 16 + i + ((lane < 16) ? 0 : 8);
                float val = acc[mi][ni][i] + bv;
                if (!split) {
                    ((float*)Outp)[(size_t)row * N + col] = val;
                } else {
                    int b = row >> 10, s = row & (Sc - 1);
                    int h = col >> 6,  d = col & (DHc - 1);
                    ((__bf16*)Outp)[((((size_t)b * Hc + h) * Sc + s) << 6) + d] = (__bf16)val;
                }
            }
        }
    }
}

// ---------------------------------------------------------------------------
// Flash attention: one workgroup per (64-query block, (b,h)); 4 waves x 16 rows.
// ---------------------------------------------------------------------------
__global__ __launch_bounds__(128)
void attn_flash_wmma(const __bf16* __restrict__ qh,
                     const __bf16* __restrict__ kh,
                     const __bf16* __restrict__ vh,
                     const unsigned char* __restrict__ mask,
                     const unsigned char* __restrict__ semask,
                     __bf16* __restrict__ ctx) {
    __shared__ __bf16 Ks[64][72];        // [key][dh], +8 pad
    __shared__ __bf16 Vt[64][72];        // [dh][key]
    __shared__ __bf16 Ps[4][16][72];     // per-wave P: C-layout -> A-layout bounce

    const int tid   = threadIdx.x;
    const int lane  = tid & 31;
    const int wave  = tid >> 5;
    const int r15   = lane & 15;
    const int kb    = (lane < 16) ? 0 : 8;
    const int bh    = blockIdx.y;
    const int b     = bh >> 4;
    const int h     = bh & 15;
    const int qbase = blockIdx.x * 64;

    const __bf16* Qp = qh + (size_t)bh * Sc * DHc;
    const __bf16* Kp = kh + (size_t)bh * Sc * DHc;
    const __bf16* Vp = vh + (size_t)bh * Sc * DHc;

    v16bf aq[2];
    {
        const __bf16* qrow = Qp + (size_t)(qbase + wave * 16 + r15) * DHc;
#pragma unroll
        for (int f = 0; f < 2; ++f) {
            v8bf lo = *(const v8bf*)(qrow + f * 32 + kb);
            v8bf hi = *(const v8bf*)(qrow + f * 32 + kb + 16);
#pragma unroll
            for (int i = 0; i < 8; ++i) { aq[f][i] = lo[i]; aq[f][i + 8] = hi[i]; }
        }
    }

    float m_i[8], l_i[8];
#pragma unroll
    for (int i = 0; i < 8; ++i) { m_i[i] = -3.0e38f; l_i[i] = 0.0f; }
    v8f o[4];
#pragma unroll
    for (int nd = 0; nd < 4; ++nd) o[nd] = zero8();

    for (int kt = 0; kt < Sc / 64; ++kt) {
        const int k0 = kt * 64;

        // ---- stage K tile: TDM -> async -> plain; V tile: register transpose ----
#if HAVE_TDM
        if (tid == 0) {
            // 64x64 bf16 tile, row stride 64; LDS rows: 64 elems = 32 DW -> code 4;
            // pad 8 elems = 4 DW -> code 3.
            tdm_load_2d_bf16(Kp + (size_t)k0 * DHc, lds_offset(&Ks[0][0]),
                             64u, 64u, (unsigned)DHc, (unsigned)DHc, (unsigned)Sc,
                             4, 3);
        }
#elif HAVE_ASYNC
#pragma unroll
        for (int it = 0; it < 4; ++it) {
            int idx = it * 128 + tid;
            int row = idx >> 3, c8 = idx & 7;
            __builtin_amdgcn_global_load_async_to_lds_b128(
                (const __attribute__((address_space(1))) void*)
                    (Kp + (size_t)(k0 + row) * DHc + c8 * 8),
                (__attribute__((address_space(3))) void*)&Ks[row][c8 * 8], 0, 0);
        }
#else
#pragma unroll
        for (int it = 0; it < 4; ++it) {
            int idx = it * 128 + tid;
            int row = idx >> 3, c8 = idx & 7;
            *(v8bf*)&Ks[row][c8 * 8] =
                *(const v8bf*)(Kp + (size_t)(k0 + row) * DHc + c8 * 8);
        }
#endif
#pragma unroll
        for (int it = 0; it < 4; ++it) {
            int idx = it * 128 + tid;
            int row = idx >> 3, c8 = idx & 7;
            v8bf vv = *(const v8bf*)(Vp + (size_t)(k0 + row) * DHc + c8 * 8);
#pragma unroll
            for (int j = 0; j < 8; ++j) Vt[c8 * 8 + j][row] = vv[j];
        }
#if HAVE_TDM
        if (tid == 0) __builtin_amdgcn_s_wait_tensorcnt(0);
#elif HAVE_ASYNC
#if __has_builtin(__builtin_amdgcn_s_wait_asynccnt)
        __builtin_amdgcn_s_wait_asynccnt(0);
#else
        asm volatile("s_wait_asynccnt 0" ::: "memory");
#endif
#endif
        __syncthreads();

        // ---- S = Q K^T ----
        v8f sc[4];
#pragma unroll
        for (int nk = 0; nk < 4; ++nk) {
            v16bf b0 = ld_frag(&Ks[nk * 16 + r15][0], kb);
            v16bf b1 = ld_frag(&Ks[nk * 16 + r15][32], kb);
            v8f z = zero8();
            z = __builtin_amdgcn_wmma_f32_16x16x32_bf16(false, aq[0], false, b0,
                                                        (short)0, z, false, false);
            sc[nk] = __builtin_amdgcn_wmma_f32_16x16x32_bf16(false, aq[1], false, b1,
                                                             (short)0, z, false, false);
        }

        // ---- scale + key-padding masks + online softmax ----
        float p[4][8];
#pragma unroll
        for (int nk = 0; nk < 4; ++nk) {
            int col = k0 + nk * 16 + r15;
            bool mk = (mask[(size_t)b * Sc + col] | semask[(size_t)b * Sc + col]) != 0;
#pragma unroll
            for (int i = 0; i < 8; ++i)
                p[nk][i] = mk ? -1.0e9f : sc[nk][i] * 0.125f;
        }
#pragma unroll
        for (int i = 0; i < 8; ++i) {
            float mx = fmaxf(fmaxf(p[0][i], p[1][i]), fmaxf(p[2][i], p[3][i]));
#pragma unroll
            for (int off = 8; off >= 1; off >>= 1)
                mx = fmaxf(mx, __shfl_xor(mx, off, 32));
            float mnew  = fmaxf(m_i[i], mx);
            float alpha = __expf(m_i[i] - mnew);
            float rs = 0.0f;
#pragma unroll
            for (int nk = 0; nk < 4; ++nk) {
                float e = __expf(p[nk][i] - mnew);
                p[nk][i] = e;
                rs += e;
            }
#pragma unroll
            for (int off = 8; off >= 1; off >>= 1)
                rs += __shfl_xor(rs, off, 32);
            l_i[i] = l_i[i] * alpha + rs;
            m_i[i] = mnew;
#pragma unroll
            for (int nd = 0; nd < 4; ++nd) o[nd][i] *= alpha;
        }

        // ---- P -> LDS (per-wave) -> A-layout ----
#pragma unroll
        for (int nk = 0; nk < 4; ++nk)
#pragma unroll
            for (int i = 0; i < 8; ++i)
                Ps[wave][i + ((lane < 16) ? 0 : 8)][nk * 16 + r15] = (__bf16)p[nk][i];

        // ---- O += P V ----
#pragma unroll
        for (int half = 0; half < 2; ++half) {
            v16bf ap = ld_frag(&Ps[wave][r15][half * 32], kb);
#pragma unroll
            for (int nd = 0; nd < 4; ++nd) {
                v16bf bv = ld_frag(&Vt[nd * 16 + r15][half * 32], kb);
                o[nd] = __builtin_amdgcn_wmma_f32_16x16x32_bf16(false, ap, false, bv,
                                                                (short)0, o[nd], false, false);
            }
        }
        __syncthreads();
    }

#pragma unroll
    for (int nd = 0; nd < 4; ++nd) {
#pragma unroll
        for (int i = 0; i < 8; ++i) {
            int s = qbase + wave * 16 + i + ((lane < 16) ? 0 : 8);
            int d = h * DHc + nd * 16 + r15;
            ctx[((size_t)b * Sc + s) * Dc + d] = (__bf16)(o[nd][i] / l_i[i]);
        }
    }
}

// ---------------------------------------------------------------------------
extern "C" void kernel_launch(void* const* d_in, const int* in_sizes, int n_in,
                              void* d_out, int out_size, void* d_ws, size_t ws_size,
                              hipStream_t stream) {
    const float* v  = (const float*)d_in[0];
    const float* k  = (const float*)d_in[1];
    const float* q  = (const float*)d_in[2];
    const unsigned char* mask   = (const unsigned char*)d_in[3];
    const unsigned char* semask = (const unsigned char*)d_in[4];
    const float* Wv = (const float*)d_in[5];
    const float* bv = (const float*)d_in[6];
    const float* Wk = (const float*)d_in[7];
    const float* bk = (const float*)d_in[8];
    const float* Wq = (const float*)d_in[9];
    const float* bq = (const float*)d_in[10];
    const float* Wm = (const float*)d_in[11];
    const float* bm = (const float*)d_in[12];

    const size_t headElems = (size_t)Bc * Hc * Sc * DHc;
    __bf16* qh  = (__bf16*)d_ws;
    __bf16* kh  = qh + headElems;
    __bf16* vh  = kh + headElems;
    __bf16* ctx = vh + headElems;

    dim3 gGrid(Dc / 128, Mtot / 128);
    gemm128_wmma_bf16<0><<<gGrid, 256, 0, stream>>>(v, Wv, bv, vh, 1);
    gemm128_wmma_bf16<0><<<gGrid, 256, 0, stream>>>(k, Wk, bk, kh, 1);
    gemm128_wmma_bf16<0><<<gGrid, 256, 0, stream>>>(q, Wq, bq, qh, 1);

    attn_flash_wmma<<<dim3(Sc / 64, Bc * Hc), 128, 0, stream>>>(qh, kh, vh, mask,
                                                                semask, ctx);

    gemm128_wmma_bf16<1><<<gGrid, 256, 0, stream>>>(ctx, Wm, bm, (float*)d_out, 0);
}
